// JumpReLUSparseAutoencoder_45801531244816
// MI455X (gfx1250) — matrix operand order repaired and checked by reading, not compiled, not run on hardware
//
#include <hip/hip_runtime.h>
#include <stdint.h>

typedef unsigned short u16;
typedef __attribute__((ext_vector_type(16))) __bf16 v16bf;
typedef __attribute__((ext_vector_type(8)))  float  v8f;
typedef int v4i __attribute__((vector_size(16)));   // matches builtin param type

#define N_ROWS 8192
#define DM 1024
#define DS 16384
#define BM 64            // rows per block
#define XLD (DM + 8)     // Xs row stride in u16 (516 dwords -> conflict-free quads)
#define ALD 72           // As row stride in u16 (36 dwords  -> conflict-free quads)

#if defined(__has_builtin)
#  if __has_builtin(__builtin_amdgcn_global_load_async_to_lds_b128) && \
      __has_builtin(__builtin_amdgcn_s_wait_asynccnt)
#    define HAVE_ASYNC_LDS 1
#  endif
#endif

union Frag {
    v16bf  v;
    float4 f4[2];
};

__device__ __forceinline__ u16 f32_to_bf16(float f) {
    union { float f; unsigned u; } x;
    x.f = f;
    unsigned u = x.u;
    // round-to-nearest-even truncation to bf16
    unsigned r = (u + 0x7fffu + ((u >> 16) & 1u)) >> 16;
    return (u16)r;
}

// ---------------------------------------------------------------------------
// Prep kernel 1: pack a f32 weight matrix (K x N, row stride `ld` == N) into
// fragment-major bf16: tiles of 16(N) x 32(K); tile layout = contiguous-k B-operand
// VGPR order (lane L: n = L%16, k = (L/16)*16 .. +15), 32 bytes per lane,
// 1 KB per tile, tiles ordered n_tile-major then k_tile.
// ---------------------------------------------------------------------------
__global__ void pack_frags(const float* __restrict__ src, u16* __restrict__ dst,
                           int ld, int k_tiles) {
    int id   = blockIdx.x * blockDim.x + threadIdx.x;
    int lane = id & 31;
    int tile = id >> 5;
    int n_tile = tile / k_tiles;
    int k_tile = tile - n_tile * k_tiles;
    int n  = n_tile * 16 + (lane & 15);
    int kb = k_tile * 32 + (lane >> 4) * 16;
    union { u16 s[16]; float4 f4[2]; } t;
#pragma unroll
    for (int j = 0; j < 16; ++j)
        t.s[j] = f32_to_bf16(src[(size_t)(kb + j) * ld + n]);
    float4* o = (float4*)(dst + (size_t)id * 16);
    o[0] = t.f4[0];
    o[1] = t.f4[1];
}

// Prep kernel 2: straight f32 -> bf16 conversion (4 elements / thread).
__global__ void cvt_bf16_vec(const float* __restrict__ src, u16* __restrict__ dst, int n4) {
    int id = blockIdx.x * blockDim.x + threadIdx.x;
    if (id >= n4) return;
    float4 v = ((const float4*)src)[id];
    union { u16 s[4]; uint2 u; } t;
    t.s[0] = f32_to_bf16(v.x);
    t.s[1] = f32_to_bf16(v.y);
    t.s[2] = f32_to_bf16(v.z);
    t.s[3] = f32_to_bf16(v.w);
    ((uint2*)dst)[id] = t.u;
}

// ---------------------------------------------------------------------------
// Fused JumpReLU SAE: per block of 64 rows, stream D_SAE in 64-feature
// chunks:  GEMM1 (encode) -> gate -> store sae_acts + stash bf16 acts in LDS
//          GEMM2 (decode) accumulates recon in VGPRs.
// ---------------------------------------------------------------------------
__global__ __launch_bounds__(512) void sae_fused(
    const u16* __restrict__ Xb,     // [8192][1024] bf16
    const u16* __restrict__ WeF,    // W_enc fragment-major bf16 (n_tiles=1024, k_tiles=32)
    const u16* __restrict__ WdF,    // W_dec fragment-major bf16 (n_tiles=64, k_tiles=512)
    const float* __restrict__ b_enc,
    const float* __restrict__ b_dec,
    const float* __restrict__ theta,
    float* __restrict__ recon,      // [8192][1024]
    float* __restrict__ sae)        // [8192][16384]
{
    __shared__ u16 Xs[BM][XLD];     // 64 x 1032 bf16 = 129 KB
    __shared__ u16 As[BM][ALD];     // 64 x 72   bf16 =   9 KB

    const int tid  = threadIdx.x;
    const int lane = tid & 31;
    const int w    = tid >> 5;        // wave 0..15
    const int l15  = lane & 15;
    const int lhi  = lane >> 4;       // 0: lanes 0-15, 1: lanes 16-31
    const int r0   = blockIdx.x * BM;
    const int aoff0 = lhi * 16;       // A-fragment k sub-offset per half-wave

    // ---- stage X tile (64 x 1024 bf16) into LDS, 16B per transfer ----------
    {
        const u16* gsrc = Xb + (size_t)r0 * DM;
#pragma unroll
        for (int it = 0; it < 16; ++it) {
            int idx = it * 512 + tid;        // 8192 16B-chunks total
            int r = idx >> 7;                // row 0..63
            int c = idx & 127;               // 16B chunk within row
            const u16* g = gsrc + (size_t)r * DM + c * 8;
            u16* l = &Xs[r][c * 8];
#if defined(HAVE_ASYNC_LDS)
            __builtin_amdgcn_global_load_async_to_lds_b128(
                (__attribute__((address_space(1))) v4i*)(g),
                (__attribute__((address_space(3))) v4i*)(l), 0, 0);
#else
            *(float4*)l = *(const float4*)g;
#endif
        }
#if defined(HAVE_ASYNC_LDS)
        __builtin_amdgcn_s_wait_asynccnt(0);
#endif
    }
    __syncthreads();

    // GEMM1 tile of this wave: (mi, ni) within the 64x64 pre-tile
    const int mi = w >> 2;
    const int ni = w & 3;
    const u16* XsRow = &Xs[mi * 16 + l15][0];

    // recon accumulators: this wave owns columns [w*64, w*64+64)
    v8f acc2[4][4];
#pragma unroll
    for (int a = 0; a < 4; ++a)
#pragma unroll
        for (int b = 0; b < 4; ++b)
            acc2[a][b] = (v8f){0.f, 0.f, 0.f, 0.f, 0.f, 0.f, 0.f, 0.f};

    for (int chunk = 0; chunk < DS / 64; ++chunk) {
        // ---------------- GEMM1: pre = X @ W_enc (K = 1024) ----------------
        v8f c1 = (v8f){0.f, 0.f, 0.f, 0.f, 0.f, 0.f, 0.f, 0.f};
        const u16* bbase = WeF + ((size_t)(chunk * 4 + ni) * 32) * 512 + lane * 16;
#pragma unroll 8
        for (int ks = 0; ks < 32; ++ks) {
            Frag a, b;
            const u16* ar = XsRow + ks * 32 + aoff0;
            a.f4[0] = *(const float4*)(ar);
            a.f4[1] = *(const float4*)(ar + 8);
            const u16* br = bbase + (size_t)ks * 512;
            b.f4[0] = *(const float4*)(br);
            b.f4[1] = *(const float4*)(br + 8);
            c1 = __builtin_amdgcn_wmma_f32_16x16x32_bf16(
                false, a.v, false, b.v, (short)0, c1, false, false);
        }

        // ---------------- JumpReLU gate + store sae_acts -------------------
        const int nloc = ni * 16 + l15;
        const int ng   = chunk * 64 + nloc;
        const float be = b_enc[ng];
        const float th = theta[ng];
        float* srow = sae + (size_t)(r0 + mi * 16 + lhi * 8) * DS + ng;
#pragma unroll
        for (int i = 0; i < 8; ++i) {
            float v   = c1[i] + be;
            float act = v > th ? v : 0.0f;
            srow[(size_t)i * DS] = act;
            As[mi * 16 + lhi * 8 + i][nloc] = f32_to_bf16(act);
        }
        __syncthreads();

        // ---------------- GEMM2: recon += acts @ W_dec (K = 64) ------------
#pragma unroll
        for (int kst = 0; kst < 2; ++kst) {
            Frag a2[4];
#pragma unroll
            for (int m2 = 0; m2 < 4; ++m2) {
                const u16* ar = &As[m2 * 16 + l15][kst * 32 + aoff0];
                a2[m2].f4[0] = *(const float4*)(ar);
                a2[m2].f4[1] = *(const float4*)(ar + 8);
            }
#pragma unroll
            for (int n2 = 0; n2 < 4; ++n2) {
                const u16* br = WdF +
                    ((size_t)((w * 4 + n2) * 512 + chunk * 2 + kst)) * 512 + lane * 16;
                Frag b2;
                b2.f4[0] = *(const float4*)(br);
                b2.f4[1] = *(const float4*)(br + 8);
#pragma unroll
                for (int m2 = 0; m2 < 4; ++m2)
                    acc2[m2][n2] = __builtin_amdgcn_wmma_f32_16x16x32_bf16(
                        false, a2[m2].v, false, b2.v, (short)0, acc2[m2][n2],
                        false, false);
            }
        }
        __syncthreads();   // protect As before next chunk overwrites it
    }

    // ---------------- epilogue: recon = acc + b_dec ------------------------
#pragma unroll
    for (int n2 = 0; n2 < 4; ++n2) {
        const int n = w * 64 + n2 * 16 + l15;
        const float bd = b_dec[n];
#pragma unroll
        for (int m2 = 0; m2 < 4; ++m2) {
            float* rrow = recon + (size_t)(r0 + m2 * 16 + lhi * 8) * DM + n;
#pragma unroll
            for (int i = 0; i < 8; ++i)
                rrow[(size_t)i * DM] = acc2[m2][n2][i] + bd;
        }
    }
}

// ---------------------------------------------------------------------------
extern "C" void kernel_launch(void* const* d_in, const int* in_sizes, int n_in,
                              void* d_out, int out_size, void* d_ws, size_t ws_size,
                              hipStream_t stream) {
    (void)in_sizes; (void)n_in; (void)out_size; (void)ws_size;

    const float* X    = (const float*)d_in[0];   // [8192,1024]
    const float* Wenc = (const float*)d_in[1];   // [1024,16384]
    const float* Wdec = (const float*)d_in[2];   // [16384,1024]
    const float* benc = (const float*)d_in[3];   // [16384]
    const float* bdec = (const float*)d_in[4];   // [1024]
    const float* thet = (const float*)d_in[5];   // [16384]

    float* recon = (float*)d_out;                       // [8192,1024]
    float* sae   = recon + (size_t)N_ROWS * DM;         // [8192,16384]

    // workspace layout (needs 80 MB): WencF 32MB | WdecF 32MB | X bf16 16MB
    u16* wencF = (u16*)d_ws;
    u16* wdecF = wencF + (size_t)DS * DM;
    u16* xb    = wdecF + (size_t)DS * DM;

    // W_enc: K=1024, N=16384 -> n_tiles=1024, k_tiles=32 -> 1M threads
    pack_frags<<<4096, 256, 0, stream>>>(Wenc, wencF, DS, 32);
    // W_dec: K=16384, N=1024 -> n_tiles=64, k_tiles=512 -> 1M threads
    pack_frags<<<4096, 256, 0, stream>>>(Wdec, wdecF, DM, 512);
    // X -> bf16
    cvt_bf16_vec<<<(N_ROWS * DM / 4) / 256, 256, 0, stream>>>(X, xb, N_ROWS * DM / 4);

    sae_fused<<<N_ROWS / BM, 512, 0, stream>>>(xb, wencF, wdecF, benc, bdec,
                                               thet, recon, sae);
}
